// Attention_910533066913
// MI455X (gfx1250) — matrix operand order repaired and checked by reading, not compiled, not run on hardware
//
#include <hip/hip_runtime.h>
#include <hip/hip_bf16.h>
#include <math.h>

// ---- CDNA5 (gfx1250) wave32 WMMA types -------------------------------------
typedef __attribute__((ext_vector_type(16))) __bf16 v16bf;
typedef __attribute__((ext_vector_type(8)))  float  v8f;

#define B_  32
#define S_  4096
#define D_  256
#define U_  256

#define BPITCH 40   // bf16 per LDS slab row: 32 data + 8 pad (16B-aligned, bank-friendly)

// ---- workspace layout (bytes) ----------------------------------------------
//  off 0        : bias[b][u]  = pos@W1 + b1 + b2      (32 KB,  f32)
//  off 32768    : W2^T bf16   [U][D]                  (128 KB, bf16)
//  off 163840   : score[b][s]                         (512 KB, f32)
//  off 688128   : red[b][2]   = {max, sumexp}         (256 B,  f32)
//  off 688384   : context partials [b][16][d]         (512 KB, f32)
#define OFF_BIAS   0
#define OFF_W2T    32768
#define OFF_SCORE  163840
#define OFF_RED    688128
#define OFF_PART   688384

// ---------------------------------------------------------------------------
// bias[b,u] = sum_d position[b,d]*W1[d,u] + b1[u] + b2[u]
__global__ void k_bias(const float* __restrict__ position,
                       const float* __restrict__ W1,
                       const float* __restrict__ b1,
                       const float* __restrict__ b2,
                       float* __restrict__ bias) {
  const int b = blockIdx.x, u = threadIdx.x;
  const float* pos = position + b * D_;
  float acc = b1[u] + b2[u];
  #pragma unroll 4
  for (int d = 0; d < D_; ++d) acc = fmaf(pos[d], W1[d * U_ + u], acc);
  bias[b * U_ + u] = acc;
}

// W2^T in bf16: w2t[u*D + d] = bf16(W2[d*U + u])  -> per-lane-contiguous B frags
__global__ void k_w2t(const float* __restrict__ W2, __bf16* __restrict__ w2t) {
  const int u = blockIdx.x, d = threadIdx.x;
  w2t[u * D_ + d] = (__bf16)W2[d * U_ + u];
}

// ---------------------------------------------------------------------------
// Fused: score[b,s] = tanh(options[b,s,:]@W2 + bias[b,:]) . V + bv
// 8 waves/block = 128 S-rows; B slab (256x32 bf16) double-buffered in LDS,
// loaded cooperatively once per K-step and consumed by all 8 waves.
__global__ void __launch_bounds__(256)
k_score(const float*  __restrict__ options,
        const __bf16* __restrict__ w2t,
        const float*  __restrict__ bias_all,
        const float*  __restrict__ V,
        const float*  __restrict__ bvp,
        float*        __restrict__ score) {
  __shared__ __align__(16) __bf16 bslab[2][U_ * BPITCH];   // 2 x 20 KB

  const int tid  = threadIdx.x;
  const int b    = blockIdx.y;
  const int wave = tid >> 5;
  const int lane = tid & 31;
  const int hi   = lane >> 4;     // which 16-lane half
  const int lh   = lane & 15;     // lane within half
  const int s0   = blockIdx.x * 128 + wave * 16;
  const int row  = s0 + lh;

  const float* arow = options + ((size_t)b * S_ + row) * D_;
  const float* bias = bias_all + b * U_;

  // ---- stage B slab for kt = 0 : w2t[u][0..32) -> bslab[0][u*BPITCH ..]
  #pragma unroll
  for (int i = 0; i < 4; ++i) {
    const int task = tid + i * 256;          // 1024 tasks = 256 rows x 4 segs
    const int r = task >> 2, seg = task & 3; // seg = 8 bf16 = one uint4
    *(uint4*)(&bslab[0][r * BPITCH + seg * 8]) =
        *(const uint4*)(w2t + (size_t)r * D_ + seg * 8);
  }
  __syncthreads();

  v8f acc[16] = {};

  for (int kt = 0; kt < 8; ++kt) {          // K = 256, 32 per step
    // ---- prefetch next B slab into registers (uniform branch: EXEC stays full)
    uint4 stage[4];
    if (kt < 7) {
      #pragma unroll
      for (int i = 0; i < 4; ++i) {
        const int task = tid + i * 256;
        const int r = task >> 2, seg = task & 3;
        stage[i] = *(const uint4*)(w2t + (size_t)r * D_ + (kt + 1) * 32 + seg * 8);
      }
    }

    // ---- A fragment: 16x32 bf16, ISA layout:
    //   lanes 0-15 : K = kt*32 + {0..7, 16..23}
    //   lanes16-31 : K = kt*32 + {8..15, 24..31}
    const float* ap = arow + kt * 32 + hi * 8;
    __builtin_prefetch(ap + 32, 0, 3);      // next K-tile -> global_prefetch_b8
    const float4 p0 = ((const float4*)ap)[0];
    const float4 p1 = ((const float4*)ap)[1];
    const float4 p2 = ((const float4*)(ap + 16))[0];
    const float4 p3 = ((const float4*)(ap + 16))[1];
    v16bf a;
    a[0]=(__bf16)p0.x;  a[1]=(__bf16)p0.y;  a[2] =(__bf16)p0.z;  a[3] =(__bf16)p0.w;
    a[4]=(__bf16)p1.x;  a[5]=(__bf16)p1.y;  a[6] =(__bf16)p1.z;  a[7] =(__bf16)p1.w;
    a[8]=(__bf16)p2.x;  a[9]=(__bf16)p2.y;  a[10]=(__bf16)p2.z;  a[11]=(__bf16)p2.w;
    a[12]=(__bf16)p3.x; a[13]=(__bf16)p3.y; a[14]=(__bf16)p3.z;  a[15]=(__bf16)p3.w;

    // ---- B fragments from LDS: lane = column, K = {hi*8..hi*8+7, 16+hi*8..16+hi*8+7}
    const __bf16* bbuf = &bslab[kt & 1][lh * BPITCH + hi * 8];
    #pragma unroll
    for (int ut = 0; ut < 16; ++ut) {
      union { uint4 q[2]; v16bf v; } bb;
      bb.q[0] = *(const uint4*)(bbuf + ut * 16 * BPITCH);
      bb.q[1] = *(const uint4*)(bbuf + ut * 16 * BPITCH + 16);
      acc[ut] = __builtin_amdgcn_wmma_f32_16x16x32_bf16(
          /*neg_a=*/false, a, /*neg_b=*/false, bb.v,
          /*c_mod=*/(short)0, acc[ut], /*reuse_a=*/false, /*reuse_b=*/false);
    }

    // ---- commit next slab to the alternate buffer
    if (kt < 7) {
      #pragma unroll
      for (int i = 0; i < 4; ++i) {
        const int task = tid + i * 256;
        const int r = task >> 2, seg = task & 3;
        *(uint4*)(&bslab[(kt + 1) & 1][r * BPITCH + seg * 8]) = stage[i];
      }
    }
    __syncthreads();
  }

  // ---- fused epilogue: tanh + dot with V, reduce across the 16-lane half
  const float bv = bvp[0];
  float part[8] = {};
  #pragma unroll
  for (int ut = 0; ut < 16; ++ut) {
    const int   u  = ut * 16 + lh;          // C/D layout: N = lane within half
    const float bu = bias[u];
    const float vu = V[u];
    #pragma unroll
    for (int j = 0; j < 8; ++j) {           // C/D layout: M = j + hi*8
      const float t = tanhf(acc[ut][j] + bu);
      part[j] = fmaf(t, vu, part[j]);
    }
  }
  #pragma unroll
  for (int j = 0; j < 8; ++j) {
    float v = part[j];
    v += __shfl_xor(v, 1, 32);
    v += __shfl_xor(v, 2, 32);
    v += __shfl_xor(v, 4, 32);
    v += __shfl_xor(v, 8, 32);              // stays within each 16-lane half
    part[j] = v;
  }
  if (lh == 0) {
    #pragma unroll
    for (int j = 0; j < 8; ++j)
      score[(size_t)b * S_ + s0 + hi * 8 + j] = part[j] + bv;
  }
}

// ---------------------------------------------------------------------------
// Per-batch softmax statistics over S
__global__ void k_stats(const float* __restrict__ score, float* __restrict__ red) {
  const int b = blockIdx.x, tid = threadIdx.x;
  __shared__ float sm[256];
  float m = -INFINITY;
  for (int s = tid; s < S_; s += 256) m = fmaxf(m, score[(size_t)b * S_ + s]);
  sm[tid] = m; __syncthreads();
  for (int off = 128; off > 0; off >>= 1) {
    if (tid < off) sm[tid] = fmaxf(sm[tid], sm[tid + off]);
    __syncthreads();
  }
  m = sm[0]; __syncthreads();
  float sum = 0.f;
  for (int s = tid; s < S_; s += 256) sum += __expf(score[(size_t)b * S_ + s] - m);
  sm[tid] = sum; __syncthreads();
  for (int off = 128; off > 0; off >>= 1) {
    if (tid < off) sm[tid] += sm[tid + off];
    __syncthreads();
  }
  if (tid == 0) { red[b * 2] = m; red[b * 2 + 1] = sm[0]; }
}

// weights[b,s] -> d_out (second output region)
__global__ void k_weights(const float* __restrict__ score,
                          const float* __restrict__ red,
                          float* __restrict__ wout) {
  const int b = blockIdx.y;
  const int s = blockIdx.x * 256 + threadIdx.x;
  const float m = red[b * 2], inv = 1.f / red[b * 2 + 1];
  wout[(size_t)b * S_ + s] = __expf(score[(size_t)b * S_ + s] - m) * inv;
}

// context partials: 16 S-chunks per batch in parallel
__global__ void k_cpart(const float* __restrict__ options,
                        const float* __restrict__ wts,
                        float* __restrict__ part) {
  const int b = blockIdx.x, c = blockIdx.y, d = threadIdx.x;
  const float* opt = options + ((size_t)b * S_ + (size_t)c * 256) * D_ + d;
  const float* w   = wts + (size_t)b * S_ + c * 256;
  float acc = 0.f;
  #pragma unroll 4
  for (int s = 0; s < 256; ++s) acc = fmaf(w[s], opt[(size_t)s * D_], acc);
  part[((size_t)b * 16 + c) * D_ + d] = acc;
}

__global__ void k_creduce(const float* __restrict__ part, float* __restrict__ ctx) {
  const int b = blockIdx.x, d = threadIdx.x;
  float acc = 0.f;
  #pragma unroll
  for (int c = 0; c < 16; ++c) acc += part[((size_t)b * 16 + c) * D_ + d];
  ctx[b * D_ + d] = acc;
}

// ---------------------------------------------------------------------------
extern "C" void kernel_launch(void* const* d_in, const int* in_sizes, int n_in,
                              void* d_out, int out_size, void* d_ws, size_t ws_size,
                              hipStream_t stream) {
  (void)in_sizes; (void)n_in; (void)out_size; (void)ws_size;
  const float* position = (const float*)d_in[0];   // [B,D]
  const float* options  = (const float*)d_in[1];   // [B,S,D]
  const float* W1       = (const float*)d_in[2];   // [D,U]
  const float* b1       = (const float*)d_in[3];   // [U]
  const float* W2       = (const float*)d_in[4];   // [D,U]
  const float* b2       = (const float*)d_in[5];   // [U]
  const float* V        = (const float*)d_in[6];   // [U,1]
  const float* bv       = (const float*)d_in[7];   // [1]

  char* ws = (char*)d_ws;
  float*  ws_bias  = (float*)(ws + OFF_BIAS);
  __bf16* ws_w2t   = (__bf16*)(ws + OFF_W2T);
  float*  ws_score = (float*)(ws + OFF_SCORE);
  float*  ws_red   = (float*)(ws + OFF_RED);
  float*  ws_part  = (float*)(ws + OFF_PART);

  float* out_ctx = (float*)d_out;            // [B,D]
  float* out_w   = out_ctx + B_ * D_;        // [B,S,1]

  k_bias   <<<dim3(B_),         dim3(U_),  0, stream>>>(position, W1, b1, b2, ws_bias);
  k_w2t    <<<dim3(U_),         dim3(D_),  0, stream>>>(W2, ws_w2t);
  k_score  <<<dim3(S_/128, B_), dim3(256), 0, stream>>>(options, ws_w2t, ws_bias, V, bv, ws_score);
  k_stats  <<<dim3(B_),         dim3(256), 0, stream>>>(ws_score, ws_red);
  k_weights<<<dim3(S_/256, B_), dim3(256), 0, stream>>>(ws_score, ws_red, out_w);
  k_cpart  <<<dim3(B_, 16),     dim3(D_),  0, stream>>>(options, out_w, ws_part);
  k_creduce<<<dim3(B_),         dim3(D_),  0, stream>>>(ws_part, out_ctx);
}
